// PipeDecoderBlock_28381143892656
// MI455X (gfx1250) — matrix operand order repaired and checked by reading, not compiled, not run on hardware
//
#include <hip/hip_runtime.h>
#include <math.h>

// ---------------------------------------------------------------------------
// MI455X (gfx1250): wave32, WMMA bf16 16x16x32 f32-accum implicit GEMMs.
// NHWC activations so every LDS/global staging op is b128; weight tiles are
// copied global->LDS with CDNA5 async-load instructions (ASYNCcnt path).
// ---------------------------------------------------------------------------

typedef __bf16 bf16;
typedef bf16  v16bf __attribute__((ext_vector_type(16)));
typedef bf16  v8bf  __attribute__((ext_vector_type(8)));
typedef bf16  v4bf  __attribute__((ext_vector_type(4)));
typedef float v8f   __attribute__((ext_vector_type(8)));

#define HW   128
#define HW2  16384        // 128*128

// CDNA5 async global->LDS 128-bit copy (tracked by ASYNCcnt).
// VDST holds the wave-relative LDS byte address (= low 32 bits of the generic
// shared-memory pointer), VADDR the 64-bit global address.
__device__ __forceinline__ void async_copy_b128(const void* g, void* l) {
  asm volatile("global_load_async_to_lds_b128 %0, %1, off"
               :: "v"((unsigned)(uintptr_t)l), "v"(g) : "memory");
}
__device__ __forceinline__ void async_wait() {
  asm volatile("s_wait_asynccnt 0x0" ::: "memory");
}

// A-fragment: 16x32 (MxK) bf16 from LDS tile (row stride 32).
__device__ __forceinline__ v16bf load_a_frag(const bf16* As, int lane) {
  const bf16* row = As + (lane & 15) * 32 + ((lane >> 4) << 3);
  v8bf lo = *(const v8bf*)(row);
  v8bf hi = *(const v8bf*)(row + 16);
  v16bf r;
#pragma unroll
  for (int i = 0; i < 8; ++i) { r[i] = lo[i]; r[i + 8] = hi[i]; }
  return r;
}
// B-fragment: 32x16 (KxN) bf16, LDS stored [n][k] (row stride 32).
__device__ __forceinline__ v16bf load_b_frag(const bf16* Bs, int lane) {
  const bf16* row = Bs + (lane & 15) * 32 + ((lane >> 4) << 3);
  v8bf lo = *(const v8bf*)(row);
  v8bf hi = *(const v8bf*)(row + 16);
  v16bf r;
#pragma unroll
  for (int i = 0; i < 8; ++i) { r[i] = lo[i]; r[i + 8] = hi[i]; }
  return r;
}
__device__ __forceinline__ v8f wmma_bf16(v16bf a, v16bf b, v8f c) {
  return __builtin_amdgcn_wmma_f32_16x16x32_bf16(false, a, false, b, (short)0, c, false, false);
}

// ---------------------------------------------------------------------------
// K0a: pack all weights.
//   wupP [tap][oc][ic]                 bf16  (conv_transpose per-parity GEMM B)
//   wrefP[o][k = tap*256 + c]          bf16  (3x3 conv, tap-major K)
//   wdefP[o][k = tap*128 + c]          bf16  (deform GEMM, tap-major K)
//   whP  [c][t][ch], wvP [c][t][ch]    f32   (1x15 / 15x1 offset convs)
//   wmP  [c][tap][ch]                  f32   (3x3 mask conv)
// ---------------------------------------------------------------------------
__global__ void k_pack_w(const float* __restrict__ w_up, const float* __restrict__ w_ref,
                         const float* __restrict__ w_def, const float* __restrict__ w_offh,
                         const float* __restrict__ w_offv, const float* __restrict__ w_mask,
                         bf16* __restrict__ wupP, bf16* __restrict__ wrefP,
                         bf16* __restrict__ wdefP, float* __restrict__ whP,
                         float* __restrict__ wvP, float* __restrict__ wmP) {
  int i = blockIdx.x * 256 + threadIdx.x;
  if (i < 131072) {            // w_up (ic256, oc128, 2, 2)
    int tx = i & 1, ty = (i >> 1) & 1, oc = (i >> 2) & 127, ic = i >> 9;
    wupP[((ty * 2 + tx) * 128 + oc) * 256 + ic] = (bf16)w_up[i];
  }
  if (i < 294912) {            // w_ref (o128, c256, 3, 3) -> tap-major
    int kx = i % 3, ky = (i / 3) % 3, c = (i / 9) % 256, o = i / 2304;
    wrefP[o * 2304 + (ky * 3 + kx) * 256 + c] = (bf16)w_ref[i];
  }
  if (i < 147456) {            // w_def (o128, c128, 3, 3) -> tap-major
    int kx = i % 3, ky = (i / 3) % 3, c = (i / 9) % 128, o = i / 1152;
    wdefP[o * 1152 + (ky * 3 + kx) * 128 + c] = (bf16)w_def[i];
  }
  if (i < 34560) {             // w_offh/w_offv (18, 128, 15) -> [c][t][ch]
    int t = i % 15, ch = (i / 15) % 128, c = i / 1920;
    whP[(c * 15 + t) * 128 + ch] = w_offh[i];
    wvP[(c * 15 + t) * 128 + ch] = w_offv[i];
  }
  if (i < 10368) {             // w_mask (9, 128, 3, 3) -> [c][tap][ch]
    int kx = i % 3, ky = (i / 3) % 3, ch = (i / 9) % 128, c = i / 1152;
    wmP[(c * 9 + ky * 3 + kx) * 128 + ch] = w_mask[i];
  }
}

// K0b: x (B,256,64,64) NCHW f32 -> xT NHWC bf16 [b][iy][ix][ic].
__global__ void k_pack_x(const float* __restrict__ x, bf16* __restrict__ xT) {
  int j = blockIdx.x * 256 + threadIdx.x;
  if (j >= 2 * 4096 * 256) return;
  int ic = j & 255, p = j >> 8;
  int b = p >> 12, iyix = p & 4095;
  xT[j] = (bf16)x[((b * 256 + ic) << 12) + iyix];
}

// ---------------------------------------------------------------------------
// K1: conv_transpose 2x2 s2 as 4 parity GEMMs. cat is NHWC bf16 (B,H,W,256),
// channels 0..127 written here (+bias), 128..255 = skip (K1b).
// ---------------------------------------------------------------------------
__global__ void __launch_bounds__(256) k_upsample(const bf16* __restrict__ xT,
                                                  const bf16* __restrict__ wupP,
                                                  const float* __restrict__ b_up,
                                                  bf16* __restrict__ cat) {
  __shared__ bf16 As[32 * 32];
  __shared__ bf16 Bs[128 * 32];
  const int tid = threadIdx.x, lane = tid & 31, wv = tid >> 5;
  const int tile = blockIdx.x, tap = blockIdx.y, b = blockIdx.z;
  v8f acc0 = {}, acc1 = {};
  for (int kc = 0; kc < 8; ++kc) {
    const int k0 = kc * 32;
    if (tid < 128) {                              // A: 32 px x 32 ic, b128
      int m = tid >> 2, q = tid & 3;
      int p = tile * 32 + m;
      *(v8bf*)(As + m * 32 + q * 8) =
          *(const v8bf*)(xT + ((size_t)(b * 4096 + p)) * 256 + k0 + q * 8);
    }
#pragma unroll
    for (int it = 0; it < 2; ++it) {              // B: 128 n x 32 k, async b128
      int idx = it * 256 + tid;
      int n = idx >> 2, q = idx & 3;
      async_copy_b128(wupP + (tap * 128 + n) * 256 + k0 + q * 8, Bs + n * 32 + q * 8);
    }
    async_wait();
    __syncthreads();
    v16bf bf = load_b_frag(Bs + wv * 16 * 32, lane);
    v16bf a0 = load_a_frag(As, lane);
    v16bf a1 = load_a_frag(As + 16 * 32, lane);
    acc0 = wmma_bf16(a0, bf, acc0);
    acc1 = wmma_bf16(a1, bf, acc1);
    __syncthreads();
  }
  const int n = wv * 16 + (lane & 15);
  const int py = tap >> 1, px = tap & 1;
  const float bias = b_up[n];
#pragma unroll
  for (int r = 0; r < 8; ++r) {
    int mf = r + ((lane >> 4) << 3);
    int p = tile * 32 + mf, iy = p >> 6, ix = p & 63;
    cat[((size_t)(b * HW2 + (2 * iy + py) * HW + 2 * ix + px)) * 256 + n] = (bf16)(acc0[r] + bias);
    p = tile * 32 + 16 + mf; iy = p >> 6; ix = p & 63;
    cat[((size_t)(b * HW2 + (2 * iy + py) * HW + 2 * ix + px)) * 256 + n] = (bf16)(acc1[r] + bias);
  }
}

// K1b: skip (B,128,H,W) NCHW f32 -> cat NHWC channels 128..255.
__global__ void k_skip_to_cat(const float* __restrict__ skip, bf16* __restrict__ cat) {
  int i = blockIdx.x * 256 + threadIdx.x;
  if (i >= 2 * HW2 * 128) return;
  int c = i & 127, pix = i >> 7;
  int b = pix >> 14, yx = pix & 16383;
  cat[(size_t)pix * 256 + 128 + c] = (bf16)skip[((b * 128 + c) << 14) + yx];
}

// ---------------------------------------------------------------------------
// K2: 3x3 conv (256ch cat -> 128ch) implicit GEMM, K=2304 tap-major.
// Each 32-wide K-chunk = one tap over 32 contiguous channels -> b128 staging.
// conv1 is NHWC f32.
// ---------------------------------------------------------------------------
__global__ void __launch_bounds__(256) k_conv3x3(const bf16* __restrict__ cat,
                                                 const bf16* __restrict__ wrefP,
                                                 const float* __restrict__ b_ref,
                                                 float* __restrict__ conv1) {
  __shared__ bf16 As[32 * 32];
  __shared__ bf16 Bs[128 * 32];
  const int tid = threadIdx.x, lane = tid & 31, wv = tid >> 5;
  const int base = blockIdx.x * 32;          // 32 pixels, same row (128/32)
  const int b = base >> 14, yx0 = base & 16383;
  const int y = yx0 >> 7, x0 = yx0 & 127;
  v8f acc0 = {}, acc1 = {};
  for (int kc = 0; kc < 72; ++kc) {
    const int k0 = kc * 32;
    const int kid = k0 >> 8, c0 = k0 & 255;
    const int dy = kid / 3 - 1, dx = kid % 3 - 1;
    if (tid < 128) {                              // A: 32 px x 32 c, b128
      int m = tid >> 2, q = tid & 3;
      int sy = y + dy, sx = x0 + m + dx;
      v8bf v = {};
      if (sy >= 0 && sy < HW && sx >= 0 && sx < HW)
        v = *(const v8bf*)(cat + ((size_t)(b * HW2 + (sy << 7) + sx)) * 256 + c0 + q * 8);
      *(v8bf*)(As + m * 32 + q * 8) = v;
    }
#pragma unroll
    for (int it = 0; it < 2; ++it) {              // B: async b128
      int idx = it * 256 + tid;
      int n = idx >> 2, q = idx & 3;
      async_copy_b128(wrefP + n * 2304 + k0 + q * 8, Bs + n * 32 + q * 8);
    }
    async_wait();
    __syncthreads();
    v16bf bf = load_b_frag(Bs + wv * 16 * 32, lane);
    v16bf a0 = load_a_frag(As, lane);
    v16bf a1 = load_a_frag(As + 16 * 32, lane);
    acc0 = wmma_bf16(a0, bf, acc0);
    acc1 = wmma_bf16(a1, bf, acc1);
    __syncthreads();
  }
  const int n = wv * 16 + (lane & 15);
  const float bias = b_ref[n];
#pragma unroll
  for (int r = 0; r < 8; ++r) {
    int mf = r + ((lane >> 4) << 3);
    conv1[((size_t)(base + mf)) * 128 + n]      = acc0[r] + bias;
    conv1[((size_t)(base + 16 + mf)) * 128 + n] = acc1[r] + bias;
  }
}

// ---------------------------------------------------------------------------
// BN stats on NHWC tensor, deterministic 2-stage reduction.
// stage1: 256 blocks x 128 pixels -> part[blk][c] (sum) / part[32768+...] (sq)
// ---------------------------------------------------------------------------
__global__ void k_bnstats1(const float* __restrict__ t, float* __restrict__ part) {
  __shared__ float s1[256], s2[256];
  const int blk = blockIdx.x, tid = threadIdx.x;
  const int c = tid & 127, half = tid >> 7;
  const float* base = t + (size_t)blk * 128 * 128;
  float s = 0.f, q = 0.f;
  for (int p = half * 64; p < half * 64 + 64; ++p) {
    float v = base[p * 128 + c];
    s += v; q += v * v;
  }
  s1[tid] = s; s2[tid] = q;
  __syncthreads();
  if (tid < 128) {
    part[blk * 128 + tid]         = s1[tid] + s1[tid + 128];
    part[32768 + blk * 128 + tid] = s2[tid] + s2[tid + 128];
  }
}
__global__ void k_bnstats2(const float* __restrict__ part, float* __restrict__ stats) {
  int c = threadIdx.x;   // 128 threads
  float s = 0.f, q = 0.f;
  for (int b = 0; b < 256; ++b) { s += part[b * 128 + c]; q += part[32768 + b * 128 + c]; }
  float m = s * (1.f / 32768.f);
  stats[c] = m;
  stats[128 + c] = q * (1.f / 32768.f) - m * m;
}

// BN+ReLU, NHWC in -> NHWC out.
__global__ void k_bnrelu(const float* __restrict__ in, const float* __restrict__ stats,
                         const float* __restrict__ g, const float* __restrict__ be,
                         float* __restrict__ out) {
  int i = blockIdx.x * 256 + threadIdx.x;
  if (i >= 2 * HW2 * 128) return;
  int c = i & 127;
  float r = (in[i] - stats[c]) * rsqrtf(stats[128 + c] + 1e-5f) * g[c] + be[c];
  out[i] = fmaxf(r, 0.f);
}
// BN+ReLU, NHWC in -> NCHW out (final output).
__global__ void k_bnrelu_nchw(const float* __restrict__ in, const float* __restrict__ stats,
                              const float* __restrict__ g, const float* __restrict__ be,
                              float* __restrict__ out) {
  int i = blockIdx.x * 256 + threadIdx.x;
  if (i >= 2 * HW2 * 128) return;
  int c = i & 127, pix = i >> 7;
  int b = pix >> 14, yx = pix & 16383;
  float r = (in[i] - stats[c]) * rsqrtf(stats[128 + c] + 1e-5f) * g[c] + be[c];
  out[((size_t)(b * 128 + c)) * HW2 + yx] = fmaxf(r, 0.f);
}

// ---------------------------------------------------------------------------
// K5: offset = conv(1x15,pad(0,7)) + conv(15x1,pad(7,0)), h NHWC, float4 dots.
// offs layout [pix][18].
// ---------------------------------------------------------------------------
__global__ void k_offset(const float* __restrict__ h, const float* __restrict__ whP,
                         const float* __restrict__ bh, const float* __restrict__ wvP,
                         const float* __restrict__ bv, float* __restrict__ offs) {
  int i = blockIdx.x * 256 + threadIdx.x;
  if (i >= 2 * HW2 * 18) return;
  int c = i % 18, pix = i / 18;
  int b = pix >> 14, yx = pix & 16383, y = yx >> 7, x = yx & 127;
  float acc = bh[c] + bv[c];
  for (int t = 0; t < 15; ++t) {
    int sx = x + t - 7;
    if (sx >= 0 && sx < HW) {
      const float4* p4 = (const float4*)(h + ((size_t)(b * HW2 + (y << 7) + sx)) * 128);
      const float4* w4 = (const float4*)(whP + (c * 15 + t) * 128);
#pragma unroll 8
      for (int j = 0; j < 32; ++j) {
        float4 a = p4[j], w = w4[j];
        acc += a.x * w.x + a.y * w.y + a.z * w.z + a.w * w.w;
      }
    }
    int sy = y + t - 7;
    if (sy >= 0 && sy < HW) {
      const float4* p4 = (const float4*)(h + ((size_t)(b * HW2 + (sy << 7) + x)) * 128);
      const float4* w4 = (const float4*)(wvP + (c * 15 + t) * 128);
#pragma unroll 8
      for (int j = 0; j < 32; ++j) {
        float4 a = p4[j], w = w4[j];
        acc += a.x * w.x + a.y * w.y + a.z * w.z + a.w * w.w;
      }
    }
  }
  offs[i] = acc;
}

// K6: mask = sigmoid(conv3x3(h,128->9,pad1)). mask layout [pix][9].
__global__ void k_maskconv(const float* __restrict__ h, const float* __restrict__ wmP,
                           const float* __restrict__ bm, float* __restrict__ maskb) {
  int i = blockIdx.x * 256 + threadIdx.x;
  if (i >= 2 * HW2 * 9) return;
  int c = i % 9, pix = i / 9;
  int b = pix >> 14, yx = pix & 16383, y = yx >> 7, x = yx & 127;
  float acc = bm[c];
  for (int kid = 0; kid < 9; ++kid) {
    int sy = y + kid / 3 - 1, sx = x + kid % 3 - 1;
    if (sy >= 0 && sy < HW && sx >= 0 && sx < HW) {
      const float4* p4 = (const float4*)(h + ((size_t)(b * HW2 + (sy << 7) + sx)) * 128);
      const float4* w4 = (const float4*)(wmP + (c * 9 + kid) * 128);
#pragma unroll 8
      for (int j = 0; j < 32; ++j) {
        float4 a = p4[j], w = w4[j];
        acc += a.x * w.x + a.y * w.y + a.z * w.z + a.w * w.w;
      }
    }
  }
  maskb[i] = 1.f / (1.f + expf(-acc));
}

// ---------------------------------------------------------------------------
// K7: deformable 3x3 conv as WMMA GEMM, K=1152 tap-major (k = tap*128 + c).
// Per chunk: one tap, 32 contiguous channels; bilinear gathers are float4
// (4 channels share one sample point). dconv is NHWC f32.
// ---------------------------------------------------------------------------
__global__ void __launch_bounds__(256) k_deform(const float* __restrict__ h,
                                                const float* __restrict__ offs,
                                                const float* __restrict__ maskb,
                                                const bf16* __restrict__ wdefP,
                                                float* __restrict__ dconv) {
  __shared__ bf16 As[16 * 32];
  __shared__ bf16 Bs[128 * 32];
  __shared__ int   sy0[144], sx0[144];
  __shared__ float swy[144], swx[144], smk[144];
  const int tid = threadIdx.x, lane = tid & 31, wv = tid >> 5;
  const int base = blockIdx.x * 16;
  const int b = base >> 14;
  if (tid < 144) {
    int m = tid / 9, kk = tid - m * 9;
    int P = base + m, yx = P & 16383, y = yx >> 7, x = yx & 127;
    float dy = offs[(size_t)P * 18 + 2 * kk];
    float dx = offs[(size_t)P * 18 + 2 * kk + 1];
    float py = dy + (float)(y + kk / 3 - 1);
    float px = dx + (float)(x + kk % 3 - 1);
    float fy = floorf(py), fx = floorf(px);
    sy0[tid] = (int)fy; sx0[tid] = (int)fx;
    swy[tid] = py - fy;  swx[tid] = px - fx;
    smk[tid] = maskb[(size_t)P * 9 + kk];
  }
  __syncthreads();
  v8f acc = {};
  for (int kc = 0; kc < 36; ++kc) {
    const int k0 = kc * 32;
    const int kid = k0 >> 7, c0 = k0 & 127;
    if (tid < 128) {                 // A: 16 px x 32 c, float4-sampled
      int m = tid >> 3, g = tid & 7;
      int e = m * 9 + kid;
      int y0 = sy0[e], x1 = sx0[e];
      float wy = swy[e], wx = swx[e], mk = smk[e];
      const float* hb = h + ((size_t)(b * HW2)) * 128 + c0 + g * 4;
      float4 z = {0.f, 0.f, 0.f, 0.f};
      bool yi0 = (unsigned)y0 < 128u, yi1 = (unsigned)(y0 + 1) < 128u;
      bool xi0 = (unsigned)x1 < 128u, xi1 = (unsigned)(x1 + 1) < 128u;
      float4 v00 = (yi0 && xi0) ? *(const float4*)(hb + ((y0 << 7) + x1) * 128) : z;
      float4 v01 = (yi0 && xi1) ? *(const float4*)(hb + ((y0 << 7) + x1 + 1) * 128) : z;
      float4 v10 = (yi1 && xi0) ? *(const float4*)(hb + (((y0 + 1) << 7) + x1) * 128) : z;
      float4 v11 = (yi1 && xi1) ? *(const float4*)(hb + (((y0 + 1) << 7) + x1 + 1) * 128) : z;
      float w00 = (1.f - wy) * (1.f - wx) * mk, w01 = (1.f - wy) * wx * mk;
      float w10 = wy * (1.f - wx) * mk,         w11 = wy * wx * mk;
      v4bf o;
      o[0] = (bf16)(v00.x * w00 + v01.x * w01 + v10.x * w10 + v11.x * w11);
      o[1] = (bf16)(v00.y * w00 + v01.y * w01 + v10.y * w10 + v11.y * w11);
      o[2] = (bf16)(v00.z * w00 + v01.z * w01 + v10.z * w10 + v11.z * w11);
      o[3] = (bf16)(v00.w * w00 + v01.w * w01 + v10.w * w10 + v11.w * w11);
      *(v4bf*)(As + m * 32 + g * 4) = o;
    }
#pragma unroll
    for (int it = 0; it < 2; ++it) {  // B: async b128
      int idx = it * 256 + tid;
      int n = idx >> 2, q = idx & 3;
      async_copy_b128(wdefP + n * 1152 + k0 + q * 8, Bs + n * 32 + q * 8);
    }
    async_wait();
    __syncthreads();
    v16bf a  = load_a_frag(As, lane);
    v16bf bf = load_b_frag(Bs + wv * 16 * 32, lane);
    acc = wmma_bf16(a, bf, acc);
    __syncthreads();
  }
  const int n = wv * 16 + (lane & 15);
#pragma unroll
  for (int r = 0; r < 8; ++r) {
    int mf = r + ((lane >> 4) << 3);
    dconv[((size_t)(base + mf)) * 128 + n] = acc[r];
  }
}

// ---------------------------------------------------------------------------
// Host orchestration. Workspace layout (bytes):
//  cat(bf16 NHWC)@0 16M | conv1/dconv(f32 NHWC)@16M 16M | h(f32 NHWC)@32M 16M
//  offs@48M 2.25M | mask@51M 1.13M | part@53M 256K | stats@53M+256K
//  xT(bf16)@54M 4M | packed weights @58M (~1.5M)
// ---------------------------------------------------------------------------
extern "C" void kernel_launch(void* const* d_in, const int* in_sizes, int n_in,
                              void* d_out, int out_size, void* d_ws, size_t ws_size,
                              hipStream_t stream) {
  const float* x      = (const float*)d_in[0];
  const float* skip   = (const float*)d_in[1];
  const float* w_up   = (const float*)d_in[2];
  const float* b_up   = (const float*)d_in[3];
  const float* w_ref  = (const float*)d_in[4];
  const float* b_ref  = (const float*)d_in[5];
  const float* g1     = (const float*)d_in[6];
  const float* be1    = (const float*)d_in[7];
  const float* w_offh = (const float*)d_in[8];
  const float* b_offh = (const float*)d_in[9];
  const float* w_offv = (const float*)d_in[10];
  const float* b_offv = (const float*)d_in[11];
  const float* w_mask = (const float*)d_in[12];
  const float* b_mask = (const float*)d_in[13];
  const float* w_def  = (const float*)d_in[14];
  const float* g2     = (const float*)d_in[15];
  const float* be2    = (const float*)d_in[16];

  char*  ws     = (char*)d_ws;
  bf16*  cat    = (bf16*)(ws);
  float* conv1  = (float*)(ws + ((size_t)16 << 20));
  float* h      = (float*)(ws + ((size_t)32 << 20));
  float* offs   = (float*)(ws + ((size_t)48 << 20));
  float* maskb  = (float*)(ws + ((size_t)51 << 20));
  float* part   = (float*)(ws + ((size_t)53 << 20));
  float* stats1 = (float*)(ws + ((size_t)53 << 20) + (256 << 10));
  float* stats2 = stats1 + 256;
  bf16*  xT     = (bf16*)(ws + ((size_t)54 << 20));
  char*  wbase  = ws + ((size_t)58 << 20);
  bf16*  wupP   = (bf16*)(wbase);                    // 262144 B
  bf16*  wrefP  = (bf16*)(wbase + 262144);           // 589824 B
  bf16*  wdefP  = (bf16*)(wbase + 262144 + 589824);  // 294912 B
  float* whP    = (float*)(wbase + 1146880);         // 138240 B
  float* wvP    = (float*)(wbase + 1146880 + 138240);
  float* wmP    = (float*)(wbase + 1146880 + 2 * 138240);
  float* dconv  = conv1;   // conv1 dead once h exists

  const int ELTS = 2 * HW2 * 128;  // 4,194,304

  k_pack_w<<<(294912 + 255) / 256, 256, 0, stream>>>(w_up, w_ref, w_def, w_offh, w_offv,
                                                     w_mask, wupP, wrefP, wdefP, whP, wvP, wmP);
  k_pack_x<<<(2097152 + 255) / 256, 256, 0, stream>>>(x, xT);
  k_upsample<<<dim3(128, 4, 2), 256, 0, stream>>>(xT, wupP, b_up, cat);
  k_skip_to_cat<<<(ELTS + 255) / 256, 256, 0, stream>>>(skip, cat);
  k_conv3x3<<<1024, 256, 0, stream>>>(cat, wrefP, b_ref, conv1);
  k_bnstats1<<<256, 256, 0, stream>>>(conv1, part);
  k_bnstats2<<<1, 128, 0, stream>>>(part, stats1);
  k_bnrelu<<<(ELTS + 255) / 256, 256, 0, stream>>>(conv1, stats1, g1, be1, h);
  k_offset<<<(2 * HW2 * 18 + 255) / 256, 256, 0, stream>>>(h, whP, b_offh, wvP, b_offv, offs);
  k_maskconv<<<(2 * HW2 * 9 + 255) / 256, 256, 0, stream>>>(h, wmP, b_mask, maskb);
  k_deform<<<2048, 256, 0, stream>>>(h, offs, maskb, wdefP, dconv);
  k_bnstats1<<<256, 256, 0, stream>>>(dconv, part);
  k_bnstats2<<<1, 128, 0, stream>>>(part, stats2);
  k_bnrelu_nchw<<<(ELTS + 255) / 256, 256, 0, stream>>>(dconv, stats2, g2, be2, (float*)d_out);

  (void)in_sizes; (void)n_in; (void)out_size; (void)ws_size;
}
